// TransformerBlock_79860621902048
// MI455X (gfx1250) — matrix operand order repaired and checked by reading, not compiled, not run on hardware
//
#include <hip/hip_runtime.h>
#include <cstdint>
#include <cstddef>

// ---------------------------------------------------------------------------
// CDNA5 WMMA types (wave32, v_wmma_f32_16x16x32_bf16)
// ---------------------------------------------------------------------------
typedef __bf16 bf16;
typedef __attribute__((ext_vector_type(16))) __bf16 v16bf;
typedef __attribute__((ext_vector_type(8)))  float  v8f;

union FragBf {
  v16bf v;
  uint4 q[2];
  __bf16 e[16];
};

static __device__ __forceinline__ float gelu_exact(float x) {
  // torch default GELU: 0.5*x*(1+erf(x/sqrt(2)))
  return 0.5f * x * (1.0f + erff(x * 0.7071067811865475f));
}

// A fragment (16x32 bf16 of row-major [M,K] source).
// Layout used here (contiguous k, same as B): lane row=lane&15, hi=lane>>4; elem e -> K = hi*16+e
// -> two 16B contiguous loads per lane.
static __device__ __forceinline__ v16bf load_fragA(const bf16* __restrict__ base,
                                                   int ld, int row, int k0, int hi) {
  FragBf f;
  const uint4* p = (const uint4*)(base + (long)row * ld + k0 + hi * 16);
  f.q[0] = p[0];
  f.q[1] = p[1];
  return f.v;
}

// B fragment for NT access: B[k,n] = Src[n,k], Src row-major [N,K].
// Layout used here (contiguous k, same as A): lane holds column N; K = hi*16 + e -> one 32B contiguous run.
static __device__ __forceinline__ v16bf load_fragB(const bf16* __restrict__ base,
                                                   int ld, int col, int k0, int hi) {
  FragBf f;
  const uint4* p = (const uint4*)(base + (long)col * ld + k0 + hi * 16);
  f.q[0] = p[0];
  f.q[1] = p[1];
  return f.v;
}

// Same, but f32 source with on-the-fly bf16 conversion (4x float4 loads).
static __device__ __forceinline__ v16bf load_fragB(const float* __restrict__ base,
                                                   int ld, int col, int k0, int hi) {
  const float4* p = (const float4*)(base + (long)col * ld + k0 + hi * 16);
  FragBf f;
#pragma unroll
  for (int j = 0; j < 4; ++j) {
    float4 v = p[j];
    f.e[j * 4 + 0] = (__bf16)v.x;
    f.e[j * 4 + 1] = (__bf16)v.y;
    f.e[j * 4 + 2] = (__bf16)v.z;
    f.e[j * 4 + 3] = (__bf16)v.w;
  }
  return f.v;
}

static __device__ __forceinline__ void store_vec8(bf16* p, const float* v) {
  union { uint4 q; __bf16 e[8]; } u;
#pragma unroll
  for (int r = 0; r < 8; ++r) u.e[r] = (__bf16)v[r];
  *(uint4*)p = u.q;
}
static __device__ __forceinline__ void store_vec8(float* p, const float* v) {
  float4 lo{v[0], v[1], v[2], v[3]};
  float4 hi4{v[4], v[5], v[6], v[7]};
  ((float4*)p)[0] = lo;
  ((float4*)p)[1] = hi4;
}

// ---------------------------------------------------------------------------
// Unified NT GEMM: C[b,m,n] = sum_k A[b,m,k] * B[b,n,k]
//   A: bf16 row-major [M,K] (per-batch stride sA; 0 => shared weights)
//   B: TB row-major [N,K]; rows < ksplit of k come from x0, rest from x1
//   Epilogue: +bias[m], exact GELU, +resid (NHWC [b,N,M] f32)
//   Output: OUT_MN ? row-major [b,M,N] : NHWC [b,N,M] (16B/32B vector stores)
// Wave tile 32x64 (8 wmma / 6 frag loads per K-chunk), block 64x256 (8 waves),
// double-buffered fragments (K % 64 == 0) to overlap loads with WMMA.
// ---------------------------------------------------------------------------
template <typename TB, typename TOUT, bool BIAS, bool GELU_, bool RESID, bool OUT_MN>
__global__ __launch_bounds__(256) void nt_gemm_kernel(
    const bf16* __restrict__ A, int lda, long sA,
    const TB* __restrict__ x0, const TB* __restrict__ x1,
    int ldb, long sB0, long sB1, int ksplit,
    const float* __restrict__ bias, const float* __restrict__ resid,
    TOUT* __restrict__ y, int M, int Ntot, int K) {
  const int lane = threadIdx.x & 31, wave = threadIdx.x >> 5;
  const int wm = wave >> 2, wn = wave & 3;
  const int row = lane & 15, hi = lane >> 4;
  const int b = blockIdx.z;
  const int m0 = blockIdx.y * 64 + wm * 32;
  const int n0 = blockIdx.x * 256 + wn * 64;
  const bf16* Ab = A + (long)b * sA;
  const TB* B0 = x0 + (long)b * sB0;
  const TB* B1 = x1 + (long)b * sB1;

  v8f acc[2][4];
#pragma unroll
  for (int a = 0; a < 2; ++a)
#pragma unroll
    for (int t = 0; t < 4; ++t) acc[a][t] = v8f{0.f};

  auto load_chunk = [&](int k0, v16bf (&af)[2], v16bf (&bf)[4]) {
    af[0] = load_fragA(Ab, lda, m0 + row, k0, hi);
    af[1] = load_fragA(Ab, lda, m0 + 16 + row, k0, hi);
    const TB* bb;
    int kk;
    if (k0 < ksplit) { bb = B0; kk = k0; }
    else             { bb = B1; kk = k0 - ksplit; }
#pragma unroll
    for (int t = 0; t < 4; ++t)
      bf[t] = load_fragB(bb, ldb, n0 + t * 16 + row, kk, hi);
  };
  auto mma_chunk = [&](const v16bf (&af)[2], const v16bf (&bf)[4]) {
#pragma unroll
    for (int a = 0; a < 2; ++a)
#pragma unroll
      for (int t = 0; t < 4; ++t)
        acc[a][t] = __builtin_amdgcn_wmma_f32_16x16x32_bf16(
            false, af[a], false, bf[t], (short)0, acc[a][t], false, false);
  };

  v16bf a0[2], b0[4], a1[2], b1[4];
  load_chunk(0, a0, b0);
  for (int k0 = 0; k0 < K; k0 += 64) {
    load_chunk(k0 + 32, a1, b1);     // prefetch odd chunk
    mma_chunk(a0, b0);
    if (k0 + 64 < K) load_chunk(k0 + 64, a0, b0);  // prefetch next even chunk
    mma_chunk(a1, b1);
  }

#pragma unroll
  for (int a = 0; a < 2; ++a) {
    const int mbase = m0 + a * 16 + hi * 8;
#pragma unroll
    for (int t = 0; t < 4; ++t) {
      const long n = n0 + t * 16 + row;
      float v8[8];
#pragma unroll
      for (int r = 0; r < 8; ++r) {
        float v = acc[a][t][r];
        if (BIAS) v += bias[mbase + r];
        if (GELU_) v = gelu_exact(v);
        if (RESID) v += resid[((long)b * Ntot + n) * M + mbase + r];
        v8[r] = v;
      }
      if (OUT_MN) {
#pragma unroll
        for (int r = 0; r < 8; ++r)
          y[((long)b * M + mbase + r) * Ntot + n] = (TOUT)v8[r];
      } else {
        store_vec8(y + ((long)b * Ntot + n) * M + mbase, v8);
      }
    }
  }
}

// ---------------------------------------------------------------------------
// NCHW f32 -> NHWC bf16 input transpose: in [B,64,HW] -> out [B,HW,64]
// ---------------------------------------------------------------------------
__global__ __launch_bounds__(256) void nchw_to_nhwc_bf16_kernel(
    const float* __restrict__ in, bf16* __restrict__ out, long total) {
  long i = (long)blockIdx.x * 256 + threadIdx.x;
  if (i >= total) return;
  const int p = (int)(i & 65535);
  const int c = (int)((i >> 16) & 63);
  const long b = i >> 22;
  out[((b << 16) + p) * 64 + c] = (__bf16)in[i];
}

// ---------------------------------------------------------------------------
// Depthwise 3x3 SAME, NHWC bf16: x,y [B,HW,CN]; wd [CN,1,3,3] f32
// ---------------------------------------------------------------------------
template <int CN>
__global__ __launch_bounds__(256) void dw3x3_nhwc_kernel(
    const bf16* __restrict__ x, const float* __restrict__ wd,
    bf16* __restrict__ y, long total) {
  long i = (long)blockIdx.x * 256 + threadIdx.x;
  if (i >= total) return;
  const int c = (int)(i % CN);
  const long pi = i / CN;
  const int p = (int)(pi & 65535);
  const long bbase = pi - p;  // b * HW
  const int h = p >> 8, w = p & 255;
  float s = 0.f;
#pragma unroll
  for (int di = -1; di <= 1; ++di) {
#pragma unroll
    for (int dj = -1; dj <= 1; ++dj) {
      const int h2 = h + di, w2 = w + dj;
      if (h2 >= 0 && h2 < 256 && w2 >= 0 && w2 < 256)
        s += (float)x[(bbase + h2 * 256 + w2) * CN + c] *
             wd[c * 9 + (di + 1) * 3 + (dj + 1)];
    }
  }
  y[i] = (__bf16)s;
}

// ---------------------------------------------------------------------------
// Patchify (k=8) + L2 norm + bf16.  src NHWC [B,HW,nchTot], channels [c0,c0+64)
// dst: TRANS ? [B,D,L] (for V) : [B,L,D].  One block per (b,l), D=4096, L=1024.
// ---------------------------------------------------------------------------
template <bool TRANS>
__global__ __launch_bounds__(256) void patchify_l2norm_kernel(
    const bf16* __restrict__ src, int nchTot, int c0, bf16* __restrict__ dst) {
  const int bl = blockIdx.x;
  const int b = bl >> 10, l = bl & 1023;
  const int hp = l >> 5, wp = l & 31;
  const int t = threadIdx.x;
  float xs[16];
  float ss = 0.f;
#pragma unroll
  for (int i = 0; i < 16; ++i) {
    const int d = t + i * 256;
    const int c = d >> 6, rem = d & 63, ii = rem >> 3, jj = rem & 7;
    const long pix = ((long)b << 16) + (hp * 8 + ii) * 256 + (wp * 8 + jj);
    const float v = (float)src[pix * nchTot + c0 + c];
    xs[i] = v;
    ss += v * v;
  }
  __shared__ float red[256];
  red[t] = ss;
  __syncthreads();
  for (int s = 128; s > 0; s >>= 1) {
    if (t < s) red[t] += red[t + s];
    __syncthreads();
  }
  const float inv = 1.0f / fmaxf(sqrtf(red[0]), 1e-12f);
#pragma unroll
  for (int i = 0; i < 16; ++i) {
    const int d = t + i * 256;
    const bf16 q = (__bf16)(xs[i] * inv);
    if (TRANS) dst[(((long)b << 12) + d) * 1024 + l] = q;
    else       dst[((long)bl << 12) + d] = q;
  }
}

// ---------------------------------------------------------------------------
// Row softmax: P[row,:] = softmax(S[row,:]*scale), row length 1024, bf16 out
// ---------------------------------------------------------------------------
__global__ __launch_bounds__(256) void softmax_rows_kernel(
    const float* __restrict__ S, bf16* __restrict__ P, float scale) {
  const long rowbase = (long)blockIdx.x * 1024;
  const int t = threadIdx.x;
  float vals[4];
  float mx = -3.0e38f;
#pragma unroll
  for (int i = 0; i < 4; ++i) {
    vals[i] = S[rowbase + t + i * 256] * scale;
    mx = fmaxf(mx, vals[i]);
  }
  __shared__ float red[256];
  red[t] = mx;
  __syncthreads();
  for (int s = 128; s > 0; s >>= 1) {
    if (t < s) red[t] = fmaxf(red[t], red[t + s]);
    __syncthreads();
  }
  mx = red[0];
  __syncthreads();
  float sum = 0.f;
#pragma unroll
  for (int i = 0; i < 4; ++i) {
    vals[i] = __expf(vals[i] - mx);
    sum += vals[i];
  }
  red[t] = sum;
  __syncthreads();
  for (int s = 128; s > 0; s >>= 1) {
    if (t < s) red[t] += red[t + s];
    __syncthreads();
  }
  const float inv = 1.0f / red[0];
#pragma unroll
  for (int i = 0; i < 4; ++i)
    P[rowbase + t + i * 256] = (__bf16)(vals[i] * inv);
}

// ---------------------------------------------------------------------------
// Unpatchify three attention outputs [B,L,D] + concat -> folded NHWC [B,HW,192]
// ---------------------------------------------------------------------------
__global__ __launch_bounds__(256) void unpatchify_concat_kernel(
    const bf16* __restrict__ o0, const bf16* __restrict__ o1,
    const bf16* __restrict__ o2, bf16* __restrict__ folded, long total) {
  long i = (long)blockIdx.x * 256 + threadIdx.x;
  if (i >= total) return;
  const int ch = (int)(i % 192);
  const long pi = i / 192;
  const int p = (int)(pi & 65535);
  const long b = pi >> 16;
  const int s = ch >> 6, c = ch & 63;
  const int h = p >> 8, w = p & 255;
  const int hp = h >> 3, ii = h & 7, wp = w >> 3, jj = w & 7;
  const long l = hp * 32 + wp;
  const long d = c * 64 + ii * 8 + jj;
  const bf16* src = (s == 0) ? o0 : (s == 1) ? o1 : o2;
  folded[i] = src[((b << 10) + l) * 4096 + d];
}

__global__ __launch_bounds__(256) void f32_to_bf16_kernel(
    const float* __restrict__ in, bf16* __restrict__ out, int n) {
  int i = blockIdx.x * 256 + threadIdx.x;
  if (i < n) out[i] = (__bf16)in[i];
}

// ---------------------------------------------------------------------------
// Host orchestration
// ---------------------------------------------------------------------------
extern "C" void kernel_launch(void* const* d_in, const int* in_sizes, int n_in,
                              void* d_out, int out_size, void* d_ws, size_t ws_size,
                              hipStream_t stream) {
  (void)in_sizes; (void)n_in; (void)out_size; (void)ws_size;
  const long HW = 65536, B = 2, L = 1024, D = 4096;

  const float* mr        = (const float*)d_in[0];
  const float* lr        = (const float*)d_in[1];
  const float* lr_fake   = (const float*)d_in[2];
  const float* w_qkv_mr  = (const float*)d_in[3];
  const float* w_dw_mr   = (const float*)d_in[4];
  const float* w_qkv_lr  = (const float*)d_in[5];
  const float* w_dw_lr   = (const float*)d_in[6];
  const float* w_qkv_fk  = (const float*)d_in[7];
  const float* w_dw_fk   = (const float*)d_in[8];
  const float* w_qmerge  = (const float*)d_in[9];
  const float* w_feedpre = (const float*)d_in[10];
  const float* w_ff1     = (const float*)d_in[11];
  const float* b_ff1     = (const float*)d_in[12];
  const float* w_ff2     = (const float*)d_in[13];
  const float* b_ff2     = (const float*)d_in[14];
  float* out             = (float*)d_out;

  // ---- workspace layout ----
  char* ws = (char*)d_ws;
  size_t off = 0;
  auto alloc = [&](size_t bytes) {
    size_t o = off;
    off += (bytes + 255) & ~(size_t)255;
    return o;
  };
  const size_t o_wb_qkv_mr  = alloc(192 * 64 * 2);
  const size_t o_wb_qkv_lr  = alloc(128 * 64 * 2);
  const size_t o_wb_qkv_fk  = alloc(192 * 64 * 2);
  const size_t o_wb_qmerge  = alloc(64 * 128 * 2);
  const size_t o_wb_feedpre = alloc(64 * 192 * 2);
  const size_t o_wb_ff1     = alloc(256 * 64 * 2);
  const size_t o_wb_ff2     = alloc(64 * 256 * 2);
  const size_t o_inT    = alloc((size_t)B * HW * 64 * 2);    // NHWC input (reused)
  const size_t o_pw_tmp = alloc((size_t)B * HW * 192 * 2);   // pw out (reused)
  const size_t o_dw_mr  = alloc((size_t)B * HW * 192 * 2);
  const size_t o_dw_lr  = alloc((size_t)B * HW * 128 * 2);
  const size_t o_dw_fk  = alloc((size_t)B * HW * 192 * 2);
  const size_t o_qm     = alloc((size_t)B * HW * 64 * 2);
  const size_t patBytes = (size_t)B * L * D * 2;
  const size_t o_Qn  = alloc(patBytes);   // [B,L,D]
  const size_t o_Kmr = alloc(patBytes);   // [B,L,D]
  const size_t o_Vmr = alloc(patBytes);   // [B,D,L] (transposed)
  const size_t o_Klr = alloc(patBytes);
  const size_t o_Vlr = alloc(patBytes);
  const size_t o_Kfk = alloc(patBytes);
  const size_t o_Vfk = alloc(patBytes);
  const size_t o_S   = alloc((size_t)B * L * L * 4);
  const size_t o_P   = alloc((size_t)B * L * L * 2);
  const size_t o_O0  = alloc(patBytes);
  const size_t o_O1  = alloc(patBytes);
  const size_t o_O2  = alloc(patBytes);
  // Region reuse (single in-order stream): folded (50.3MB) overlays Qn..Vmr;
  // pre f32 (33.6MB) overlays Klr..Vlr; ff1 out (67.1MB) overlays pw_tmp+dw_mr.
  const size_t o_folded = o_Qn;
  const size_t o_pre    = o_Klr;
  const size_t o_ff1o   = o_pw_tmp;

  bf16* wb_qkv_mr  = (bf16*)(ws + o_wb_qkv_mr);
  bf16* wb_qkv_lr  = (bf16*)(ws + o_wb_qkv_lr);
  bf16* wb_qkv_fk  = (bf16*)(ws + o_wb_qkv_fk);
  bf16* wb_qmerge  = (bf16*)(ws + o_wb_qmerge);
  bf16* wb_feedpre = (bf16*)(ws + o_wb_feedpre);
  bf16* wb_ff1     = (bf16*)(ws + o_wb_ff1);
  bf16* wb_ff2     = (bf16*)(ws + o_wb_ff2);
  bf16* inT    = (bf16*)(ws + o_inT);
  bf16* pw_tmp = (bf16*)(ws + o_pw_tmp);
  bf16* dw_mr  = (bf16*)(ws + o_dw_mr);
  bf16* dw_lr  = (bf16*)(ws + o_dw_lr);
  bf16* dw_fk  = (bf16*)(ws + o_dw_fk);
  bf16* qm     = (bf16*)(ws + o_qm);
  bf16* Qn  = (bf16*)(ws + o_Qn);
  bf16* Kmr = (bf16*)(ws + o_Kmr);
  bf16* Vmr = (bf16*)(ws + o_Vmr);
  bf16* Klr = (bf16*)(ws + o_Klr);
  bf16* Vlr = (bf16*)(ws + o_Vlr);
  bf16* Kfk = (bf16*)(ws + o_Kfk);
  bf16* Vfk = (bf16*)(ws + o_Vfk);
  float* Sb = (float*)(ws + o_S);
  bf16* Pb  = (bf16*)(ws + o_P);
  bf16* O0  = (bf16*)(ws + o_O0);
  bf16* O1  = (bf16*)(ws + o_O1);
  bf16* O2  = (bf16*)(ws + o_O2);
  bf16* folded = (bf16*)(ws + o_folded);
  float* pre   = (float*)(ws + o_pre);
  bf16* ff1o   = (bf16*)(ws + o_ff1o);

  const dim3 blk(256, 1, 1);
  auto cvt = [&](const float* src, bf16* dst, int n) {
    f32_to_bf16_kernel<<<(n + 255) / 256, 256, 0, stream>>>(src, dst, n);
  };
  cvt(w_qkv_mr, wb_qkv_mr, 192 * 64);
  cvt(w_qkv_lr, wb_qkv_lr, 128 * 64);
  cvt(w_qkv_fk, wb_qkv_fk, 192 * 64);
  cvt(w_qmerge, wb_qmerge, 64 * 128);
  cvt(w_feedpre, wb_feedpre, 64 * 192);
  cvt(w_ff1, wb_ff1, 256 * 64);
  cvt(w_ff2, wb_ff2, 64 * 256);

  auto gemmGrid = [&](int M, int N) {
    return dim3((unsigned)(N / 256), (unsigned)(M / 64), (unsigned)B);
  };
  const long inTotal = B * 64 * HW;

  // ---- per-source: NCHW->NHWC, 1x1 conv (WMMA), depthwise 3x3 ----
  nchw_to_nhwc_bf16_kernel<<<(unsigned)(inTotal / 256), blk, 0, stream>>>(mr, inT, inTotal);
  nt_gemm_kernel<bf16, bf16, false, false, false, false>
      <<<gemmGrid(192, (int)HW), blk, 0, stream>>>(
          wb_qkv_mr, 64, 0, inT, inT, 64, HW * 64, HW * 64, 64,
          nullptr, nullptr, pw_tmp, 192, (int)HW, 64);
  dw3x3_nhwc_kernel<192><<<(unsigned)(B * HW * 192 / 256), blk, 0, stream>>>(
      pw_tmp, w_dw_mr, dw_mr, B * HW * 192);

  nchw_to_nhwc_bf16_kernel<<<(unsigned)(inTotal / 256), blk, 0, stream>>>(lr, inT, inTotal);
  nt_gemm_kernel<bf16, bf16, false, false, false, false>
      <<<gemmGrid(128, (int)HW), blk, 0, stream>>>(
          wb_qkv_lr, 64, 0, inT, inT, 64, HW * 64, HW * 64, 64,
          nullptr, nullptr, pw_tmp, 128, (int)HW, 64);
  dw3x3_nhwc_kernel<128><<<(unsigned)(B * HW * 128 / 256), blk, 0, stream>>>(
      pw_tmp, w_dw_lr, dw_lr, B * HW * 128);

  nchw_to_nhwc_bf16_kernel<<<(unsigned)(inTotal / 256), blk, 0, stream>>>(lr_fake, inT, inTotal);
  nt_gemm_kernel<bf16, bf16, false, false, false, false>
      <<<gemmGrid(192, (int)HW), blk, 0, stream>>>(
          wb_qkv_fk, 64, 0, inT, inT, 64, HW * 64, HW * 64, 64,
          nullptr, nullptr, pw_tmp, 192, (int)HW, 64);
  dw3x3_nhwc_kernel<192><<<(unsigned)(B * HW * 192 / 256), blk, 0, stream>>>(
      pw_tmp, w_dw_fk, dw_fk, B * HW * 192);

  // ---- q-merge: 1x1 conv over concat(q_mr, q_fk), K=128 split at 64 ----
  nt_gemm_kernel<bf16, bf16, false, false, false, false>
      <<<gemmGrid(64, (int)HW), blk, 0, stream>>>(
          wb_qmerge, 128, 0, dw_mr, dw_fk, 192, HW * 192, HW * 192, 64,
          nullptr, nullptr, qm, 64, (int)HW, 128);

  // ---- patchify + l2norm (K mats row-major, V mats transposed) ----
  const unsigned patBlocks = (unsigned)(B * L);
  patchify_l2norm_kernel<false><<<patBlocks, blk, 0, stream>>>(qm, 64, 0, Qn);
  patchify_l2norm_kernel<false><<<patBlocks, blk, 0, stream>>>(dw_mr, 192, 64, Kmr);
  patchify_l2norm_kernel<true ><<<patBlocks, blk, 0, stream>>>(dw_mr, 192, 128, Vmr);
  patchify_l2norm_kernel<false><<<patBlocks, blk, 0, stream>>>(dw_lr, 128, 0, Klr);
  patchify_l2norm_kernel<true ><<<patBlocks, blk, 0, stream>>>(dw_lr, 128, 64, Vlr);
  patchify_l2norm_kernel<false><<<patBlocks, blk, 0, stream>>>(dw_fk, 192, 64, Kfk);
  patchify_l2norm_kernel<true ><<<patBlocks, blk, 0, stream>>>(dw_fk, 192, 128, Vfk);

  // ---- attention x3 (shared q): S = Q.K^T ; P = softmax(S/64) ; O = P.Vt^T
  const float scale = 0.015625f;  // 1/sqrt(4096)
  const bf16* Ks[3] = {Kmr, Klr, Kfk};
  const bf16* Vs[3] = {Vmr, Vlr, Vfk};
  bf16* Os[3] = {O0, O1, O2};
  for (int a = 0; a < 3; ++a) {
    nt_gemm_kernel<bf16, float, false, false, false, true>
        <<<gemmGrid((int)L, (int)L), blk, 0, stream>>>(
            Qn, (int)D, L * D, Ks[a], Ks[a], (int)D, L * D, L * D, (int)D,
            nullptr, nullptr, Sb, (int)L, (int)L, (int)D);
    softmax_rows_kernel<<<(unsigned)(B * L), blk, 0, stream>>>(Sb, Pb, scale);
    nt_gemm_kernel<bf16, bf16, false, false, false, true>
        <<<gemmGrid((int)L, (int)D), blk, 0, stream>>>(
            Pb, (int)L, L * L, Vs[a], Vs[a], (int)L, D * L, D * L, (int)L,
            nullptr, nullptr, Os[a], (int)L, (int)D, (int)L);
  }

  // ---- unpatchify + concat -> folded NHWC [B,HW,192] ----
  unpatchify_concat_kernel<<<(unsigned)(B * HW * 192 / 256), blk, 0, stream>>>(
      O0, O1, O2, folded, B * HW * 192);

  // ---- feed-forward ----
  nt_gemm_kernel<bf16, float, false, false, false, false>     // pre (NHWC f32)
      <<<gemmGrid(64, (int)HW), blk, 0, stream>>>(
          wb_feedpre, 192, 0, folded, folded, 192, HW * 192, HW * 192, 192,
          nullptr, nullptr, pre, 64, (int)HW, 192);
  nt_gemm_kernel<float, bf16, true, true, false, false>       // gelu(pw+b1) bf16
      <<<gemmGrid(256, (int)HW), blk, 0, stream>>>(
          wb_ff1, 64, 0, pre, pre, 64, HW * 64, HW * 64, 64,
          b_ff1, nullptr, ff1o, 256, (int)HW, 64);
  nt_gemm_kernel<bf16, float, true, true, true, true>         // out NCHW f32
      <<<gemmGrid(64, (int)HW), blk, 0, stream>>>(
          wb_ff2, 256, 0, ff1o, ff1o, 256, HW * 256, HW * 256, 256,
          b_ff2, pre, out, 64, (int)HW, 256);
}
